// Model_45243185496800
// MI455X (gfx1250) — matrix operand order repaired and checked by reading, not compiled, not run on hardware
//
#include <hip/hip_runtime.h>
#include <hip/hip_bf16.h>
#include <math.h>

typedef __attribute__((ext_vector_type(16))) _Float16 v16h;
typedef __attribute__((ext_vector_type(8)))  _Float16 v8h;
typedef __attribute__((ext_vector_type(4)))  _Float16 v4h;
typedef __attribute__((ext_vector_type(8)))  float    v8f;

#define B_SZ 256
#define T_SZ 32
#define BT_SZ (B_SZ * T_SZ)      // 8192
#define CODE_NUM 4880
#define MED_NUM 1000
#define CODE_SIZE 128
#define H_SZ 128
#define OUT_SZ 4880

#define ASTRIDE 40               // LDS row stride in halves (32 + 8 pad)

// ---------------------------------------------------------------------------
// Kernel 1: BT[n][k] = tanh( sum_d emb[k][d] * W[d][n] ), zero-padded to Kpad.
// Output is the TRANSPOSED f16 B operand for the WMMA GEMM.
// ---------------------------------------------------------------------------
__global__ void embed_tanh_T(const float* __restrict__ emb,   // [K,128]
                             const float* __restrict__ W,     // [128,128]
                             _Float16* __restrict__ BT,       // [128,Kpad]
                             int K, int Kpad)
{
    int idx = blockIdx.x * blockDim.x + threadIdx.x;
    int total = 128 * Kpad;
    if (idx >= total) return;
    int n = idx / Kpad;
    int k = idx - n * Kpad;
    float v = 0.0f;
    if (k < K) {
        float s = 0.0f;
        const float* er = emb + (size_t)k * 128;
#pragma unroll 8
        for (int d = 0; d < 128; ++d) s += er[d] * W[d * 128 + n];
        v = tanhf(s);
    }
    BT[idx] = (_Float16)v;
}

// ---------------------------------------------------------------------------
// Kernel 2: WMMA GEMM  C[M,N] = A[M,K](f32) * B[K,N], B given transposed as
// BT[N][Kpad] (f16, zero padded). Block: 256 threads = 8 waves.
// Block tile: 32(M) x 128(N); wave tile: 32(M) x 16(N) -> 2 WMMA / step.
// A tile (32x32) double-buffered through LDS (f32 -> f16 inline); B fragment
// software-pipelined in ping-ponged registers (unroll-by-2, no copies, no
// VMEM wait before WMMA). Optional row-sum of A as a staging by-product;
// optional bias+sigmoid epilogue fused into the D store. Requires M % 32 == 0.
// ---------------------------------------------------------------------------
__global__ __launch_bounds__(256) void wmma_gemm_a32_bt16(
    const float*    __restrict__ A,      // [M,K]
    const _Float16* __restrict__ BT,     // [N,Kpad]
    float*          __restrict__ C,      // [M,ldc]
    float*          __restrict__ rowsum, // [M] or nullptr
    const float*    __restrict__ bias,   // [N] or nullptr
    int doSigmoid,
    int M, int N, int K, int Kpad, int ldc)
{
    __shared__ _Float16 sA[2][32 * ASTRIDE];   // 2 x 32 rows x 32 halves
    __shared__ float    sRed[256];

    const int tid  = threadIdx.x;
    const int wave = tid >> 5;           // 0..7 -> which 16-wide N tile
    const int lane = tid & 31;
    const int mBase = blockIdx.x * 32;
    const int nBase = blockIdx.y * 128 + wave * 16;
    const bool nActive = (nBase < N);    // wave-uniform

    // A-tile staging: thread loads one float4 of the 32x32 tile.
    const int sr  = tid >> 3;            // row 0..31
    const int scB = (tid & 7) * 4;       // col 0,4,...,28
    const float* aRow = A + (size_t)(mBase + sr) * K;

    // WMMA fragment indexing (wave32), this code's contiguous-k order (not the ISA table order; A and B match):
    //  A 16x32 f16: lane l (l<16): row l,    K in {0..15}
    //               lane l (>=16): row l-16, K in {16..31}
    //  B 32x16 f16: lane l: col l&15, K in {0..15} (l<16) or {16..31} (l>=16)
    const int fr  = lane & 15;
    const int aK0 = (lane < 16) ? 0 : 16;
    const int bK0 = (lane < 16) ? 0 : 16;
    const _Float16* bPtr = BT + (size_t)(nBase + fr) * Kpad + bK0;

    v8f acc0 = {};
    v8f acc1 = {};
    float lsum = 0.0f;

    const int steps = (K + 31) >> 5;

    auto load_step = [&](int s) -> float4 {
        const int kBase = s << 5;
        float4 r;
        if (kBase + 32 <= K) {               // uniform fast path, 16B aligned
            r = *(const float4*)(aRow + kBase + scB);
        } else {                             // uniform tail path (once)
            const int c0 = kBase + scB;
            r.x = (c0     < K) ? aRow[c0]     : 0.0f;
            r.y = (c0 + 1 < K) ? aRow[c0 + 1] : 0.0f;
            r.z = (c0 + 2 < K) ? aRow[c0 + 2] : 0.0f;
            r.w = (c0 + 3 < K) ? aRow[c0 + 3] : 0.0f;
        }
        return r;
    };
    auto store_step = [&](int s, float4 r) {
        lsum += (r.x + r.y) + (r.z + r.w);
        v4h p = { (_Float16)r.x, (_Float16)r.y, (_Float16)r.z, (_Float16)r.w };
        *(v4h*)&sA[s & 1][sr * ASTRIDE + scB] = p;   // ds_store_b64
    };
    auto load_b = [&](int s) -> v16h {
        return *(const v16h*)(bPtr + (s << 5));      // 16 contiguous halves
    };
    auto compute = [&](int s, v16h b) {
        // issue all DS loads, then both WMMAs (b already in registers)
        const _Float16* ap0 = &sA[s & 1][fr * ASTRIDE + aK0];
        const _Float16* ap1 = &sA[s & 1][(16 + fr) * ASTRIDE + aK0];
        v8h x0 = *(const v8h*)ap0;
        v8h y0 = *(const v8h*)(ap0 + 8);
        v8h x1 = *(const v8h*)ap1;
        v8h y1 = *(const v8h*)(ap1 + 8);
        v16h a0 = __builtin_shufflevector(x0, y0,
                     0,1,2,3,4,5,6,7,8,9,10,11,12,13,14,15);
        v16h a1 = __builtin_shufflevector(x1, y1,
                     0,1,2,3,4,5,6,7,8,9,10,11,12,13,14,15);
        acc0 = __builtin_amdgcn_wmma_f32_16x16x32_f16(
                   false, a0, false, b, (short)0, acc0, false, false);
        acc1 = __builtin_amdgcn_wmma_f32_16x16x32_f16(
                   false, a1, false, b, (short)0, acc1, false, false);
    };

    // prologue: stage A step 0, B fragment step 0 in registers
    v16h rb0 = {};
    v16h rb1 = {};
    if (nActive) rb0 = load_b(0);
    store_step(0, load_step(0));

    // main loop: unroll-by-2 with ping-ponged B registers (no rotation copies)
    int s = 0;
    for (;;) {
        // ---- even phase: consume rb0, preload rb1 ----
        __syncthreads();                     // buf[s&1] ready; other buf free
        {
            const bool more = (s + 1 < steps);
            float4 rn;
            if (more) {
                rn = load_step(s + 1);
                if (nActive) rb1 = load_b(s + 1);
            }
            if (nActive) compute(s, rb0);
            if (more) store_step(s + 1, rn);
        }
        if (++s >= steps) break;

        // ---- odd phase: consume rb1, preload rb0 ----
        __syncthreads();
        {
            const bool more = (s + 1 < steps);
            float4 rn;
            if (more) {
                rn = load_step(s + 1);
                if (nActive) rb0 = load_b(s + 1);
            }
            if (nActive) compute(s, rb1);
            if (more) store_step(s + 1, rn);
        }
        if (++s >= steps) break;
    }

    // ---- D tiles (+ optional bias/sigmoid): lane = col lane&15,
    //      rows v + 8*(lane>=16) ----
    if (nActive) {
        const int n = nBase + (lane & 15);
        const int mOff = (lane >> 4) << 3;
        const float bv = bias ? bias[n] : 0.0f;
#pragma unroll
        for (int v = 0; v < 8; ++v) {
            float x = acc0[v] + bv;
            if (doSigmoid) x = 1.0f / (1.0f + __expf(-x));
            C[(size_t)(mBase + v + mOff) * ldc + n] = x;
        }
#pragma unroll
        for (int v = 0; v < 8; ++v) {
            float x = acc1[v] + bv;
            if (doSigmoid) x = 1.0f / (1.0f + __expf(-x));
            C[(size_t)(mBase + 16 + v + mOff) * ldc + n] = x;
        }
    }

    // ---- row-sum reduction (8 threads per row staged the same row) ----
    if (rowsum) {
        sRed[tid] = lsum;
        __syncthreads();
        if (tid < 32) {
            float s2 = 0.0f;
#pragma unroll
            for (int j = 0; j < 8; ++j) s2 += sRed[tid * 8 + j];
            rowsum[mBase + tid] = s2;
        }
    }
}

// ---------------------------------------------------------------------------
// Kernel 3: per-batch attention scores + masked softmax + context vector.
// One block per b (256 threads). Wa cached in LDS (48KB).
// ---------------------------------------------------------------------------
__global__ __launch_bounds__(256) void attn_ctx(
    const float* __restrict__ sum_c,  // [8192,128]
    const float* __restrict__ sum_m,  // [8192,128]
    const float* __restrict__ rs_c,   // [8192]
    const float* __restrict__ rs_m,   // [8192]
    const int*   __restrict__ lens,   // [256]
    const float* __restrict__ Wa,     // [384,32]
    const float* __restrict__ ba,     // [32]
    const float* __restrict__ va,     // [32]
    float*       __restrict__ ctx)    // [256,384]
{
    __shared__ float sWa[384 * 32];   // 48 KB
    __shared__ float sRep[384];
    __shared__ float sScore[32];
    __shared__ float sPart[32];
    __shared__ float sAttn[32];

    const int b   = blockIdx.x;
    const int tid = threadIdx.x;
    for (int i = tid; i < 384 * 32; i += 256) sWa[i] = Wa[i];
    __syncthreads();

    const int L = lens[b];

    for (int t = 0; t < T_SZ; ++t) {
        const int bt = b * T_SZ + t;
        if (tid < 128) {
            float ic = 1.0f / fmaxf(rs_c[bt], 1.0f);
            float im = 1.0f / fmaxf(rs_m[bt], 1.0f);
            float hc = sum_c[(size_t)bt * 128 + tid] * ic;
            float hm = sum_m[(size_t)bt * 128 + tid] * im;
            sRep[tid]       = hc;
            sRep[128 + tid] = hm;
            sRep[256 + tid] = hc * hm;
        }
        __syncthreads();
        if (tid < 32) {
            float s = ba[tid];
#pragma unroll 4
            for (int d = 0; d < 384; ++d) s += sRep[d] * sWa[d * 32 + tid];
            sPart[tid] = tanhf(s) * va[tid];
        }
        __syncthreads();
        if (tid == 0) {
            float s = 0.0f;
#pragma unroll
            for (int j = 0; j < 32; ++j) s += sPart[j];
            sScore[t] = s;
        }
        __syncthreads();
    }

    if (tid == 0) {
        float mx = -1e30f;
        for (int t = 0; t < T_SZ; ++t) {
            float s = (t < L) ? sScore[t] : -1e9f;
            mx = fmaxf(mx, s);
        }
        float den = 0.0f;
        for (int t = 0; t < T_SZ; ++t) {
            float e = __expf(((t < L) ? sScore[t] : -1e9f) - mx);
            sAttn[t] = e;
            den += e;
        }
        float inv = 1.0f / den;
        for (int t = 0; t < T_SZ; ++t) sAttn[t] *= inv;
    }
    __syncthreads();

    for (int d = tid; d < 384; d += 256) {
        const int h = d & 127;
        float acc = 0.0f;
        for (int t = 0; t < T_SZ; ++t) {
            const int bt = b * T_SZ + t;
            float ic = 1.0f / fmaxf(rs_c[bt], 1.0f);
            float im = 1.0f / fmaxf(rs_m[bt], 1.0f);
            float hc = sum_c[(size_t)bt * 128 + h] * ic;
            float hm = sum_m[(size_t)bt * 128 + h] * im;
            float r  = (d < 128) ? hc : ((d < 256) ? hm : hc * hm);
            acc += sAttn[t] * r;
        }
        ctx[(size_t)b * 384 + d] = acc;
    }
}

// ---------------------------------------------------------------------------
// Kernel 4: transpose+convert classifier weights to f16 WT[n][k].
// ---------------------------------------------------------------------------
__global__ void convert_wclsT(const float* __restrict__ W,   // [384,4880]
                              _Float16* __restrict__ WT)     // [4880,384]
{
    int idx = blockIdx.x * blockDim.x + threadIdx.x;
    if (idx >= OUT_SZ * 384) return;
    int n = idx / 384;
    int k = idx - n * 384;
    WT[idx] = (_Float16)W[(size_t)k * OUT_SZ + n];
}

// ---------------------------------------------------------------------------
extern "C" void kernel_launch(void* const* d_in, const int* in_sizes, int n_in,
                              void* d_out, int out_size, void* d_ws, size_t ws_size,
                              hipStream_t stream) {
    const float* code_x = (const float*)d_in[0];
    // d_in[1] divided, d_in[2] neighbors: unused in the reference output path
    const int*   lens   = (const int*)  d_in[3];
    const float* med    = (const float*)d_in[4];
    const float* c_emb  = (const float*)d_in[5];
    const float* m_emb  = (const float*)d_in[6];
    const float* W_c    = (const float*)d_in[7];
    const float* W_m    = (const float*)d_in[8];
    const float* Wa     = (const float*)d_in[9];
    const float* ba     = (const float*)d_in[10];
    const float* va     = (const float*)d_in[11];
    const float* W_cls  = (const float*)d_in[12];
    const float* b_cls  = (const float*)d_in[13];
    float* out = (float*)d_out;

    const int KpadC = 4896;   // ceil(4880/32)*32
    const int KpadM = 1024;   // ceil(1000/32)*32

    char* ws = (char*)d_ws;
    size_t off = 0;
    auto alloc = [&](size_t bytes) -> void* {
        void* p = ws + off;
        off += (bytes + 255) & ~(size_t)255;
        return p;
    };
    _Float16* HcT   = (_Float16*)alloc((size_t)128 * KpadC * 2);     // 1.25 MB
    _Float16* HmT   = (_Float16*)alloc((size_t)128 * KpadM * 2);     // 0.25 MB
    float*    sum_c = (float*)   alloc((size_t)BT_SZ * 128 * 4);     // 4 MB
    float*    sum_m = (float*)   alloc((size_t)BT_SZ * 128 * 4);     // 4 MB
    float*    rs_c  = (float*)   alloc((size_t)BT_SZ * 4);
    float*    rs_m  = (float*)   alloc((size_t)BT_SZ * 4);
    float*    ctxb  = (float*)   alloc((size_t)B_SZ * 384 * 4);
    _Float16* WT    = (_Float16*)alloc((size_t)OUT_SZ * 384 * 2);    // 3.75 MB
    (void)ws_size; (void)in_sizes; (void)n_in; (void)out_size;

    // 1) transformed embeddings (transposed f16, zero padded)
    {
        int tot = 128 * KpadC;
        embed_tanh_T<<<(tot + 255) / 256, 256, 0, stream>>>(c_emb, W_c, HcT, CODE_NUM, KpadC);
        tot = 128 * KpadM;
        embed_tanh_T<<<(tot + 255) / 256, 256, 0, stream>>>(m_emb, W_m, HmT, MED_NUM, KpadM);
    }

    // 2) dominant GEMMs: sum_c = code_x @ Hc, sum_m = med @ Hm (+ row-sums)
    wmma_gemm_a32_bt16<<<dim3(BT_SZ / 32, 1), 256, 0, stream>>>(
        code_x, HcT, sum_c, rs_c, nullptr, 0,
        BT_SZ, H_SZ, CODE_NUM, KpadC, H_SZ);
    wmma_gemm_a32_bt16<<<dim3(BT_SZ / 32, 1), 256, 0, stream>>>(
        med, HmT, sum_m, rs_m, nullptr, 0,
        BT_SZ, H_SZ, MED_NUM, KpadM, H_SZ);

    // 3) attention scores + softmax + context
    attn_ctx<<<B_SZ, 256, 0, stream>>>(sum_c, sum_m, rs_c, rs_m, lens,
                                       Wa, ba, va, ctxb);

    // 4) classifier GEMM with fused bias+sigmoid, writes d_out directly
    {
        int tot = OUT_SZ * 384;
        convert_wclsT<<<(tot + 255) / 256, 256, 0, stream>>>(W_cls, WT);
    }
    wmma_gemm_a32_bt16<<<dim3(B_SZ / 32, (OUT_SZ + 127) / 128), 256, 0, stream>>>(
        ctxb, WT, out, nullptr, b_cls, 1,
        B_SZ, OUT_SZ, 384, 384, OUT_SZ);
}
